// DeformableConv_5652176961656
// MI455X (gfx1250) — matrix operand order repaired and checked by reading, not compiled, not run on hardware
//
#include <hip/hip_runtime.h>
#include <hip/hip_bf16.h>

// ---------- types ----------
typedef __attribute__((ext_vector_type(16))) __bf16        v16bf;
typedef __attribute__((ext_vector_type(8)))  float         v8f;
typedef __attribute__((ext_vector_type(8)))  unsigned int  v8u;

// ---------- constants ----------
// B=4, C=256, O=256, H=W=Ho=Wo=64, K=3, KK=9, Kdim = KK*C = 2304, k-chunks of 32 -> 72
#define NB      4
#define NC      256
#define NO      256
#define HW      4096
#define KDIM    2304
#define NKC     72          // 2304/32
// workspace layout (bytes)
#define WS_AMAIN   0            // 16 o_tiles * 72 kc * 32 lanes * 32B = 1179648
#define WS_AOFF    1179648      // 2 o_tiles * 72 * 1024 = 147456
#define WS_OM      1327104      // 4 * 32 * 4096 * 4 = 2097152
#define WS_IDX     3424256      // 4*9*4096 * 16 = 2359296
#define WS_WGT     5783552      // 4*9*4096 * 16 = 2359296  (end 8142848)

__device__ __forceinline__ unsigned short f2bf(float f) {
    unsigned int u = __float_as_uint(f);
    u += 0x7FFFu + ((u >> 16) & 1u);          // round-to-nearest-even
    return (unsigned short)(u >> 16);
}
__device__ __forceinline__ unsigned int pack2(float lo, float hi) {
    return (unsigned int)f2bf(lo) | ((unsigned int)f2bf(hi) << 16);
}

// A-fragment element placement for 16-bit 16x32 A (ISA 7.12.2):
// lane = m' + 16*((k'%16)/8); vgpr = (k'%8)/2 + 4*(k'/16); half = k'%2
__device__ __forceinline__ void a_frag_pos(int mp, int kp, int& lane, int& vg, int& half) {
    lane = mp + (((kp & 15) >> 3) << 4);
    vg   = ((kp & 7) >> 1) + ((kp >> 4) << 2);
    half = kp & 1;
}

// ============================================================================
// Kernel 0: reorder + bf16-convert + fragment-swizzle both weight tensors.
// K order is kk*256 + c  (so a 32-wide K chunk has constant kk).
// ============================================================================
__global__ void prep_weights(const float* __restrict__ weight,
                             const float* __restrict__ w_off,
                             unsigned short* __restrict__ Amain,
                             unsigned short* __restrict__ Aoff) {
    int t = blockIdx.x * 256 + threadIdx.x;
    if (t < NO * KDIM) {                       // main weight: O=256
        int o = t / KDIM, k = t % KDIM;
        int kk = k >> 8, c = k & 255;
        float v = weight[(o * NC + c) * 9 + kk];
        int lane, vg, half;
        a_frag_pos(o & 15, k & 31, lane, vg, half);
        int dw = (((o >> 4) * NKC + (k >> 5)) * 32 + lane) * 8 + vg;
        Amain[dw * 2 + half] = f2bf(v);
    } else {
        t -= NO * KDIM;
        if (t >= 32 * KDIM) return;            // offset weight: M=32 (27 padded)
        int o = t / KDIM, k = t % KDIM;
        int kk = k >> 8, c = k & 255;
        float v = (o < 27) ? w_off[(o * NC + c) * 9 + kk] : 0.0f;
        int lane, vg, half;
        a_frag_pos(o & 15, k & 31, lane, vg, half);
        int dw = (((o >> 4) * NKC + (k >> 5)) * 32 + lane) * 8 + vg;
        Aoff[dw * 2 + half] = f2bf(v);
    }
}

// ============================================================================
// Kernel 1: offset conv as implicit GEMM.  M=32, N=64/block, K=2304.
// B-tile (im2col of x, zero-padded 3x3 window) built in LDS in fragment order.
// Padding handled branchlessly: always load from a clamped in-bounds address,
// multiply by {0,1} validity mask -> unconditional, clause-able strided loads.
// 8 waves: wave = (m_tile in 0..1) x (n_tile in 0..3), one 16x16 acc each.
// ============================================================================
__global__ __launch_bounds__(256) void offset_gemm(const float* __restrict__ x,
                                                   const unsigned short* __restrict__ Aoff,
                                                   float* __restrict__ om) {
    __shared__ alignas(32) unsigned int ldsB[1024];    // 4 n_tiles * 32 lanes * 8 dw
    const int nb = blockIdx.x, b = blockIdx.y;
    const int tid = threadIdx.x, lane = tid & 31, wave = tid >> 5;
    const int mt = wave & 1, nt = wave >> 1;

    v8f acc;
    #pragma unroll
    for (int e = 0; e < 8; ++e) acc[e] = 0.0f;

    const int n = tid & 63, c_oct = tid >> 6;
    const int hw = nb * 64 + n, h = hw >> 6, w = hw & 63;
    unsigned int* stp = &ldsB[(n >> 4) * 256 + (((c_oct & 1) << 4) + (n & 15)) * 8 + ((c_oct >> 1) << 2)];
    const v8u* gA = (const v8u*)Aoff;
    const float* xb = x + (size_t)b * NC * HW;

    for (int kk = 0; kk < 9; ++kk) {
        const int kh = kk / 3, kw = kk % 3;
        const int y = h + kh - 1, xq = w + kw - 1;
        const bool valid = (y >= 0) & (y < 64) & (xq >= 0) & (xq < 64);
        const float vm = valid ? 1.0f : 0.0f;      // zero-pad as a multiply
        const int pix = valid ? (y * 64 + xq) : 0; // always in-bounds address
        #pragma unroll 1
        for (int cc = 0; cc < 8; ++cc) {
            const int kc = kk * 8 + cc;
            __syncthreads();
            const int cbase = cc * 32 + c_oct * 8;
            const float* p = xb + (size_t)cbase * HW + pix;
            float v0 = p[0 * HW], v1 = p[1 * HW], v2 = p[2 * HW], v3 = p[3 * HW];
            float v4 = p[4 * HW], v5 = p[5 * HW], v6 = p[6 * HW], v7 = p[7 * HW];
            v0 *= vm; v1 *= vm; v2 *= vm; v3 *= vm;
            v4 *= vm; v5 *= vm; v6 *= vm; v7 *= vm;
            unsigned int u0 = pack2(v0, v1), u1 = pack2(v2, v3);
            unsigned int u2 = pack2(v4, v5), u3 = pack2(v6, v7);
            *(uint4*)stp = make_uint4(u0, u1, u2, u3);
            __syncthreads();
            v8u araw = gA[(mt * NKC + kc) * 32 + lane];
            v8u braw = *(const v8u*)&ldsB[nt * 256 + lane * 8];
            acc = __builtin_amdgcn_wmma_f32_16x16x32_bf16(
                false, __builtin_bit_cast(v16bf, araw),
                false, __builtin_bit_cast(v16bf, braw),
                (short)0, acc, false, false);
        }
    }
    const int hs = lane >> 4, ncol = lane & 15;
    #pragma unroll
    for (int r = 0; r < 8; ++r) {
        int orow = mt * 16 + hs * 8 + r;
        om[((size_t)(b * 32 + orow)) * HW + nb * 64 + nt * 16 + ncol] = acc[r];
    }
}

// ============================================================================
// Kernel 2: per (b,kk,hw) compute 4 clamped gather indices + 4 folded weights
// (bilinear coef * sigmoid(mask) * per-corner validity).
// ============================================================================
__global__ void bilinear_prep(const float* __restrict__ om,
                              const float* __restrict__ b_off,
                              int4* __restrict__ idxb,
                              float4* __restrict__ wgtb) {
    int t = blockIdx.x * 256 + threadIdx.x;
    if (t >= NB * 9 * HW) return;
    const int b = t / (9 * HW), r = t % (9 * HW), kk = r >> 12, hw = r & 4095;
    const float* omb = om + (size_t)b * 32 * HW;
    float dy = omb[(2 * kk) * HW + hw]     + b_off[2 * kk];
    float dx = omb[(2 * kk + 1) * HW + hw] + b_off[2 * kk + 1];
    float mv = omb[(18 + kk) * HW + hw]    + b_off[18 + kk];
    float mask = 1.0f / (1.0f + __expf(-mv));
    const int h = hw >> 6, w = hw & 63, kh = kk / 3, kw = kk % 3;
    float py = (float)(h - 1 + kh) + dy;
    float px = (float)(w - 1 + kw) + dx;
    float y0f = floorf(py), x0f = floorf(px);
    float ly = py - y0f, lx = px - x0f;
    int y0 = (int)y0f, x0 = (int)x0f, y1 = y0 + 1, x1 = x0 + 1;
    float vy0 = (y0 >= 0 && y0 < 64) ? 1.f : 0.f;
    float vy1 = (y1 >= 0 && y1 < 64) ? 1.f : 0.f;
    float vx0 = (x0 >= 0 && x0 < 64) ? 1.f : 0.f;
    float vx1 = (x1 >= 0 && x1 < 64) ? 1.f : 0.f;
    int yc0 = min(max(y0, 0), 63), yc1 = min(max(y1, 0), 63);
    int xc0 = min(max(x0, 0), 63), xc1 = min(max(x1, 0), 63);
    idxb[t] = make_int4(yc0 * 64 + xc0, yc0 * 64 + xc1, yc1 * 64 + xc0, yc1 * 64 + xc1);
    wgtb[t] = make_float4((1.f - ly) * (1.f - lx) * mask * vy0 * vx0,
                          (1.f - ly) * lx         * mask * vy0 * vx1,
                          ly         * (1.f - lx) * mask * vy1 * vx0,
                          ly         * lx         * mask * vy1 * vx1);
}

// ============================================================================
// Kernel 3: fused deformable-sample + implicit GEMM.  M=256, N=64/block.
// 8 waves = (wm 0..3: 64 rows) x (wn 0..1: 32 cols) -> 4x2 accs per wave.
// Per 32-wide K chunk (constant kk): 256 threads gather-sample 32x64 bf16
// val tile straight into fragment-ordered LDS, then 8 WMMAs per wave.
// ============================================================================
__global__ __launch_bounds__(256) void main_gemm(const float* __restrict__ x,
                                                 const unsigned short* __restrict__ Amain,
                                                 const int4* __restrict__ idxb,
                                                 const float4* __restrict__ wgtb,
                                                 const float* __restrict__ bias,
                                                 float* __restrict__ out) {
    __shared__ alignas(32) unsigned int ldsB[1024];
    const int nb = blockIdx.x, b = blockIdx.y;
    const int tid = threadIdx.x, lane = tid & 31, wave = tid >> 5;
    const int wm = wave & 3, wn = wave >> 2;

    v8f acc[4][2];
    #pragma unroll
    for (int i = 0; i < 4; ++i)
        #pragma unroll
        for (int j = 0; j < 2; ++j)
            #pragma unroll
            for (int e = 0; e < 8; ++e) acc[i][j][e] = 0.0f;

    const int n = tid & 63, c_oct = tid >> 6;
    const int hw = nb * 64 + n;
    unsigned int* stp = &ldsB[(n >> 4) * 256 + (((c_oct & 1) << 4) + (n & 15)) * 8 + ((c_oct >> 1) << 2)];
    const v8u* gA = (const v8u*)Amain;
    const float* xb = x + (size_t)b * NC * HW;

    for (int kk = 0; kk < 9; ++kk) {
        const int4   i4 = idxb[(b * 9 + kk) * HW + hw];
        const float4 w4 = wgtb[(b * 9 + kk) * HW + hw];
        #pragma unroll 1
        for (int cc = 0; cc < 8; ++cc) {
            const int kc = kk * 8 + cc;
            __syncthreads();
            const int cbase = cc * 32 + c_oct * 8;
            unsigned int u[4];
            #pragma unroll
            for (int d = 0; d < 4; ++d) {
                const float* xc0 = xb + (size_t)(cbase + 2 * d) * HW;
                const float* xc1 = xc0 + HW;
                float v0 = w4.x * xc0[i4.x] + w4.y * xc0[i4.y] + w4.z * xc0[i4.z] + w4.w * xc0[i4.w];
                float v1 = w4.x * xc1[i4.x] + w4.y * xc1[i4.y] + w4.z * xc1[i4.z] + w4.w * xc1[i4.w];
                u[d] = pack2(v0, v1);
            }
            *(uint4*)stp = make_uint4(u[0], u[1], u[2], u[3]);
            __syncthreads();
            v8u braw0 = *(const v8u*)&ldsB[(wn * 2 + 0) * 256 + lane * 8];
            v8u braw1 = *(const v8u*)&ldsB[(wn * 2 + 1) * 256 + lane * 8];
            v16bf bf0 = __builtin_bit_cast(v16bf, braw0);
            v16bf bf1 = __builtin_bit_cast(v16bf, braw1);
            #pragma unroll
            for (int i = 0; i < 4; ++i) {
                v8u araw = gA[((wm * 4 + i) * NKC + kc) * 32 + lane];
                v16bf af = __builtin_bit_cast(v16bf, araw);
                acc[i][0] = __builtin_amdgcn_wmma_f32_16x16x32_bf16(
                    false, af, false, bf0, (short)0, acc[i][0], false, false);
                acc[i][1] = __builtin_amdgcn_wmma_f32_16x16x32_bf16(
                    false, af, false, bf1, (short)0, acc[i][1], false, false);
            }
        }
    }
    const int hs = lane >> 4, ncol = lane & 15;
    #pragma unroll
    for (int i = 0; i < 4; ++i)
        #pragma unroll
        for (int j = 0; j < 2; ++j)
            #pragma unroll
            for (int r = 0; r < 8; ++r) {
                int orow = wm * 64 + i * 16 + hs * 8 + r;
                int col  = nb * 64 + (wn * 2 + j) * 16 + ncol;
                out[((size_t)(b * NO + orow)) * HW + col] = acc[i][j][r] + bias[orow];
            }
}

// ============================================================================
extern "C" void kernel_launch(void* const* d_in, const int* in_sizes, int n_in,
                              void* d_out, int out_size, void* d_ws, size_t ws_size,
                              hipStream_t stream) {
    const float* x      = (const float*)d_in[0];
    const float* w_off  = (const float*)d_in[1];
    const float* b_off  = (const float*)d_in[2];
    const float* weight = (const float*)d_in[3];
    const float* bias   = (const float*)d_in[4];
    float* out = (float*)d_out;
    char* ws = (char*)d_ws;
    unsigned short* Amain = (unsigned short*)(ws + WS_AMAIN);
    unsigned short* Aoff  = (unsigned short*)(ws + WS_AOFF);
    float*          om    = (float*)(ws + WS_OM);
    int4*           idxb  = (int4*)(ws + WS_IDX);
    float4*         wgtb  = (float4*)(ws + WS_WGT);

    prep_weights<<<(NO * KDIM + 32 * KDIM + 255) / 256, 256, 0, stream>>>(weight, w_off, Amain, Aoff);
    offset_gemm<<<dim3(HW / 64, NB), 256, 0, stream>>>(x, Aoff, om);
    bilinear_prep<<<(NB * 9 * HW + 255) / 256, 256, 0, stream>>>(om, b_off, idxb, wgtb);
    main_gemm<<<dim3(HW / 64, NB), 256, 0, stream>>>(x, Amain, idxb, wgtb, bias, out);
}
